// SAEG_XLTNet_747324309575
// MI455X (gfx1250) — compile-verified, matrix operand order checked by reading, not run on hardware
//
#include <hip/hip_runtime.h>

#define NN      512            // nodes
#define NE      16384          // edges
#define SEQ     16
#define FI      64
#define HID     128
#define FO      64
#define EMB     512
#define DFF     2048
#define ROWS    (NN * SEQ)     // 8192

typedef __attribute__((ext_vector_type(16))) _Float16     v16h;
typedef __attribute__((ext_vector_type(8)))  float        v8f;
typedef __attribute__((ext_vector_type(4)))  unsigned int u32x4;

// Load one 16x32 f16 WMMA operand fragment (A, or B with the tile stored
// transposed [col][k]) from an LDS tile stored row-major [row][k] (u16),
// starting at column offset k0.  Layout per CDNA5 ISA 7.12.2:
//   lanes 0-15 : K = k0+{0..7} in u[0..3], K = k0+{16..23} in u[4..7]
//   lanes 16-31: K = k0+{8..15},           K = k0+{24..31}
__device__ __forceinline__ v16h load_frag(const unsigned short* lds, int ld,
                                          int row, int k0, int lane) {
  union { v16h v; unsigned int u[8]; } f;
  const unsigned int* p = (const unsigned int*)(lds + row * ld);
  int kb = k0 + ((lane >> 4) << 3);
#pragma unroll
  for (int i = 0; i < 4; ++i) {
    f.u[i]     = p[(kb >> 1) + i];
    f.u[4 + i] = p[((kb + 16) >> 1) + i];
  }
  return f.v;
}

// ---------------------------------------------------------------------------
// f16-in / f32(+f16 mirror)-out strided GEMM:
//   C(m,n) = [acc*C +] sum_k Ah[m*lda+k] * Bh[n*ldb+k]  [+ bias[n]] [relu]
// Ah: [M][lda] f16 (k contiguous); Bh canonical [N][ldb] f16 (k contiguous).
// C f32 at C[m*scm+n*scn]; optional f16 mirror Ch[m*ldch+n].
// M%128==0, N%64==0, K%32==0.  256 threads = 8 waves; 128x64 tile.
// ---------------------------------------------------------------------------
__global__ __launch_bounds__(256) void gemm_wmma(
    const _Float16* __restrict__ Ah, long lda,
    const _Float16* __restrict__ Bh, long ldb,
    float* __restrict__ C, long scm, long scn,
    _Float16* __restrict__ Ch, long ldch,
    const float* __restrict__ bias,
    int K, int acc, int relu)
{
  __shared__ __attribute__((aligned(16))) unsigned short sA[128 * 32];
  __shared__ __attribute__((aligned(16))) unsigned short sB[64 * 32];
  const int tid  = threadIdx.x;
  const int lane = tid & 31;
  const int wave = tid >> 5;
  const long rbase = (long)blockIdx.y * 128;
  const long nbase = (long)blockIdx.x * 64;

  v8f c[4];
#pragma unroll
  for (int t = 0; t < 4; ++t)
#pragma unroll
    for (int j = 0; j < 8; ++j) c[t][j] = 0.0f;

  const int ar = tid >> 1;           // A tile row 0..127 (2 threads/row)
  const int ao = (tid & 1) * 16;     // 16 halves each
  const int bn = tid >> 2;           // B tile col 0..63 (4 threads/col)
  const int bo = (tid & 3) * 8;      // 8 halves each

  for (int k0 = 0; k0 < K; k0 += 32) {
    const _Float16* ap = Ah + (rbase + ar) * lda + k0 + ao;
    const _Float16* bp = Bh + (nbase + bn) * ldb + k0 + bo;
    *(u32x4*)(sA + ar * 32 + ao)     = *(const u32x4*)(ap);
    *(u32x4*)(sA + ar * 32 + ao + 8) = *(const u32x4*)(ap + 8);
    *(u32x4*)(sB + bn * 32 + bo)     = *(const u32x4*)(bp);
    if (k0 + 32 < K) {               // pull next tile toward the WGP (L2-hot)
      __builtin_prefetch(ap + 32, 0, 1);
      __builtin_prefetch(bp + 32, 0, 1);
    }
    __syncthreads();
    v16h a = load_frag(sA, 32, wave * 16 + (lane & 15), 0, lane);
#pragma unroll
    for (int t = 0; t < 4; ++t) {
      v16h b = load_frag(sB, 32, t * 16 + (lane & 15), 0, lane);
      c[t] = __builtin_amdgcn_wmma_f32_16x16x32_f16(false, a, false, b,
                                                    (short)0, c[t], false, false);
    }
    __syncthreads();
  }

#pragma unroll
  for (int t = 0; t < 4; ++t)
#pragma unroll
    for (int j = 0; j < 8; ++j) {
      long m = rbase + wave * 16 + ((lane >> 4) << 3) + j;
      long n = nbase + t * 16 + (lane & 15);
      float v = c[t][j];
      if (acc)  v += C[m * scm + n * scn];
      if (bias) v += bias[n];
      if (relu) v = fmaxf(v, 0.0f);
      C[m * scm + n * scn] = v;
      if (Ch) Ch[m * ldch + n] = (_Float16)v;
    }
}

// ---------------------------------------------------------------------------
// Flash attention, token dim L=512 (nodes), batch=(s,h), dh=64, f16 QKV input.
// qkvh: [ROWS][1536] f16, rows (l*16+s); Q col 0, K col 512, V col 1024.
// Writes f32 out + f16 mirror at (l*16+s, h*64+d).
// ---------------------------------------------------------------------------
__global__ __launch_bounds__(128) void flash_attn(
    const _Float16* __restrict__ qkvh, float* __restrict__ outp,
    _Float16* __restrict__ outh, float scale)
{
  __shared__ __attribute__((aligned(16))) unsigned short qa[64 * 64];
  __shared__ __attribute__((aligned(16))) unsigned short kt[64 * 64];
  __shared__ __attribute__((aligned(16))) unsigned short vt[64 * 64];
  __shared__ __attribute__((aligned(16))) unsigned short pp[4][16 * 64];
  const int tid  = threadIdx.x;
  const int lane = tid & 31;
  const int wave = tid >> 5;
  const int lblk = blockIdx.x;
  const int h    = blockIdx.y & 7;
  const int s    = blockIdx.y >> 3;
  const unsigned short* qu = (const unsigned short*)qkvh;

#pragma unroll
  for (int i = 0; i < 4; ++i) {                    // Q tile: 16B chunks
    int ch = tid + i * 128;                        // 0..511
    int r = ch >> 3, off = (ch & 7) * 8;
    long row = (long)(lblk * 64 + r) * SEQ + s;
    *(u32x4*)(qa + r * 64 + off) = *(const u32x4*)(qkvh + row * 1536 + h * 64 + off);
  }

  v8f o[4];
  float mrow[8], lsum[8];
#pragma unroll
  for (int t = 0; t < 4; ++t)
#pragma unroll
    for (int j = 0; j < 8; ++j) o[t][j] = 0.0f;
#pragma unroll
  for (int j = 0; j < 8; ++j) { mrow[j] = -3.0e38f; lsum[j] = 0.0f; }

  for (int mt = 0; mt < 8; ++mt) {
    __syncthreads();
#pragma unroll
    for (int i = 0; i < 4; ++i) {                  // K tile: straight copy
      int ch = tid + i * 128;
      int r = ch >> 3, off = (ch & 7) * 8;
      long row = (long)(mt * 64 + r) * SEQ + s;
      *(u32x4*)(kt + r * 64 + off) = *(const u32x4*)(qkvh + row * 1536 + 512 + h * 64 + off);
    }
#pragma unroll
    for (int i = 0; i < 32; ++i) {                 // V tile: transpose on store
      int e = tid + i * 128;
      int r = e >> 6, d = e & 63;
      long row = (long)(mt * 64 + r) * SEQ + s;
      vt[d * 64 + r] = qu[row * 1536 + 1024 + h * 64 + d];
    }
    __syncthreads();

    v8f sacc[4];
#pragma unroll
    for (int t = 0; t < 4; ++t)
#pragma unroll
      for (int j = 0; j < 8; ++j) sacc[t][j] = 0.0f;

#pragma unroll
    for (int k0 = 0; k0 < 64; k0 += 32) {
      v16h a = load_frag(qa, 64, wave * 16 + (lane & 15), k0, lane);
#pragma unroll
      for (int t = 0; t < 4; ++t) {
        v16h b = load_frag(kt, 64, t * 16 + (lane & 15), k0, lane);
        sacc[t] = __builtin_amdgcn_wmma_f32_16x16x32_f16(false, a, false, b,
                                                 (short)0, sacc[t], false, false);
      }
    }
#pragma unroll
    for (int t = 0; t < 4; ++t)
#pragma unroll
      for (int j = 0; j < 8; ++j) sacc[t][j] *= scale;

    // online softmax: accumulator VGPR j holds one row across a 16-lane half
#pragma unroll
    for (int j = 0; j < 8; ++j) {
      float mx = fmaxf(fmaxf(sacc[0][j], sacc[1][j]), fmaxf(sacc[2][j], sacc[3][j]));
      for (int msk = 1; msk < 16; msk <<= 1) mx = fmaxf(mx, __shfl_xor(mx, msk, 32));
      float mn   = fmaxf(mrow[j], mx);
      float corr = __expf(mrow[j] - mn);
      mrow[j] = mn;
      float rs = 0.0f;
#pragma unroll
      for (int t = 0; t < 4; ++t) {
        float pv = __expf(sacc[t][j] - mn);
        sacc[t][j] = pv; rs += pv;
      }
      for (int msk = 1; msk < 16; msk <<= 1) rs += __shfl_xor(rs, msk, 32);
      lsum[j] = lsum[j] * corr + rs;
#pragma unroll
      for (int t = 0; t < 4; ++t) o[t][j] *= corr;
    }
    // re-layout P (C layout -> A fragment) through wave-private LDS
#pragma unroll
    for (int t = 0; t < 4; ++t)
#pragma unroll
      for (int j = 0; j < 8; ++j) {
        int r   = ((lane >> 4) << 3) + j;
        int col = t * 16 + (lane & 15);
        union { _Float16 hh; unsigned short u; } cv;
        cv.hh = (_Float16)sacc[t][j];
        pp[wave][r * 64 + col] = cv.u;
      }
#pragma unroll
    for (int k0 = 0; k0 < 64; k0 += 32) {
      v16h a = load_frag(pp[wave], 64, lane & 15, k0, lane);
#pragma unroll
      for (int t = 0; t < 4; ++t) {
        v16h b = load_frag(vt, 64, t * 16 + (lane & 15), k0, lane);
        o[t] = __builtin_amdgcn_wmma_f32_16x16x32_f16(false, a, false, b,
                                                      (short)0, o[t], false, false);
      }
    }
  }

#pragma unroll
  for (int t = 0; t < 4; ++t)
#pragma unroll
    for (int j = 0; j < 8; ++j) {
      int r  = ((lane >> 4) << 3) + j;
      long l = (long)lblk * 64 + wave * 16 + r;
      int d  = t * 16 + (lane & 15);
      float v = o[t][j] / lsum[j];
      long idx = (l * SEQ + s) * EMB + h * 64 + d;
      outp[idx] = v;
      outh[idx] = (_Float16)v;
    }
}

// ---------------------------------------------------------------------------
// Small VALU kernels
// ---------------------------------------------------------------------------
__global__ void zero_kernel(float* p, long n) {
  long i = (long)blockIdx.x * 256 + threadIdx.x;
  if (i < n) p[i] = 0.0f;
}
__global__ void copy2_kernel(float* df, _Float16* dh,
                             const float* sf, const _Float16* sh, long n) {
  long i = (long)blockIdx.x * 256 + threadIdx.x;
  if (i < n) { df[i] = sf[i]; dh[i] = sh[i]; }
}
__global__ void cvt_kernel(_Float16* d, const float* s, long n) {
  long i = (long)blockIdx.x * 256 + threadIdx.x;
  if (i < n) d[i] = (_Float16)s[i];
}
// transpose-convert: src [R][C] f32 -> dst [C][R] f16
__global__ void cvtT_kernel(_Float16* d, const float* s, int R, int C) {
  long i = (long)blockIdx.x * 256 + threadIdx.x;
  if (i >= (long)R * C) return;
  int r = (int)(i / C), c = (int)(i % C);
  d[(long)c * R + r] = (_Float16)s[i];
}
__global__ void count_kernel(const int* dst, float* cnt, int E) {
  int e = blockIdx.x * 256 + threadIdx.x;
  if (e < E) atomicAdd(&cnt[dst[e]], 1.0f);
}
__global__ void scatter_kernel(const float* x, const int* src, const int* dst,
                               float* sum, int E, int feat) {
  long i = (long)blockIdx.x * 256 + threadIdx.x;
  if (i >= (long)E * feat) return;
  int e = (int)(i / feat), f = (int)(i % feat);
  atomicAdd(&sum[(long)dst[e] * feat + f], x[(long)src[e] * feat + f]);
}
__global__ void mean_kernel(float* sum, _Float16* dh, const float* cnt,
                            long total, int feat) {
  long i = (long)blockIdx.x * 256 + threadIdx.x;
  if (i < total) {
    float v = sum[i] / fmaxf(cnt[i / feat], 1.0f);
    sum[i] = v;
    dh[i] = (_Float16)v;
  }
}

// fused residual-add + LayerNorm (width 512) with f32 + f16-mirror outputs
__global__ __launch_bounds__(256) void add_ln(const float* __restrict__ xa,
                                              const float* __restrict__ xb,
                                              const float* __restrict__ g,
                                              const float* __restrict__ b,
                                              float* __restrict__ y,
                                              _Float16* __restrict__ yh) {
  __shared__ float red[256];
  long row = blockIdx.x;
  int tid = threadIdx.x;
  float v0 = xa[row * 512 + tid]       + xb[row * 512 + tid];
  float v1 = xa[row * 512 + 256 + tid] + xb[row * 512 + 256 + tid];
  red[tid] = v0 + v1;
  __syncthreads();
  for (int o = 128; o > 0; o >>= 1) { if (tid < o) red[tid] += red[tid + o]; __syncthreads(); }
  float mu = red[0] * (1.0f / 512.0f);
  __syncthreads();
  float d0 = v0 - mu, d1 = v1 - mu;
  red[tid] = d0 * d0 + d1 * d1;
  __syncthreads();
  for (int o = 128; o > 0; o >>= 1) { if (tid < o) red[tid] += red[tid + o]; __syncthreads(); }
  float rs = rsqrtf(red[0] * (1.0f / 512.0f) + 1e-5f);
  float y0 = d0 * rs * g[tid]       + b[tid];
  float y1 = d1 * rs * g[256 + tid] + b[256 + tid];
  y[row * 512 + tid]        = y0;
  y[row * 512 + 256 + tid]  = y1;
  yh[row * 512 + tid]       = (_Float16)y0;
  yh[row * 512 + 256 + tid] = (_Float16)y1;
}

// sLSTM scan: T=512 sequential steps, one workgroup; emits f16 H mirror.
__global__ __launch_bounds__(256) void slstm_kernel(const float* __restrict__ x,
                                                    const float* __restrict__ W,
                                                    const float* __restrict__ Bb,
                                                    _Float16* __restrict__ H) {
  __shared__ float hS[16 * 64], cS[16 * 64], nS[16 * 64], mS[16 * 64];
  __shared__ float g[16 * 256];
  int tid = threadIdx.x;
  for (int i = tid; i < 1024; i += 256) { hS[i] = cS[i] = nS[i] = mS[i] = 0.0f; }
  __syncthreads();
  for (int t = 0; t < 512; ++t) {
    const float* xt = x + (long)t * 1024;
    for (int i = tid; i < 4096; i += 256) {       // gates [16][256]
      int s = i >> 8, col = i & 255;
      float acc = Bb[col];
      for (int j = 0; j < 64; ++j) acc += xt[s * 64 + j] * W[j * 256 + col];
      for (int j = 0; j < 64; ++j) acc += hS[s * 64 + j] * W[(64 + j) * 256 + col];
      g[i] = acc;
    }
    __syncthreads();
    for (int i = tid; i < 1024; i += 256) {
      int s = i >> 6, f = i & 63;
      float zt = tanhf(g[s * 256 + f]);
      float it = g[s * 256 + 64 + f];
      float ft = g[s * 256 + 128 + f];
      float ot = g[s * 256 + 192 + f];
      float fsig = 1.0f / (1.0f + __expf(-ft));
      float osig = 1.0f / (1.0f + __expf(-ot));
      float logsig_f = -log1pf(__expf(-ft));
      float mt = fmaxf(logsig_f + mS[i], it);
      float ip = __expf(it - mt);
      float ct = fsig * cS[i] + ip * zt;
      float nt = fsig * nS[i] + ip;
      float ht = osig * (ct / nt);
      cS[i] = ct; nS[i] = nt; mS[i] = mt; hS[i] = ht;
      H[(long)t * 1024 + i] = (_Float16)ht;
    }
    __syncthreads();
  }
}

// ---------------------------------------------------------------------------
extern "C" void kernel_launch(void* const* d_in, const int* in_sizes, int n_in,
                              void* d_out, int out_size, void* d_ws, size_t ws_size,
                              hipStream_t stream) {
  (void)in_sizes; (void)n_in; (void)out_size; (void)ws_size;
  const float* x    = (const float*)d_in[0];
  const int*   eidx = (const int*)d_in[1];
  const int* esrc = eidx;
  const int* edst = eidx + NE;
  const float* s1wl = (const float*)d_in[2];
  const float* s1wr = (const float*)d_in[3];
  const float* s1b  = (const float*)d_in[4];
  const float* s2wl = (const float*)d_in[5];
  const float* s2wr = (const float*)d_in[6];
  const float* s2b  = (const float*)d_in[7];
  const float* slW  = (const float*)d_in[8];
  const float* slB  = (const float*)d_in[9];
  const float* embw = (const float*)d_in[10];
  const float* embb = (const float*)d_in[11];
  const float* eqkvw = (const float*)d_in[12];
  const float* eqkvb = (const float*)d_in[13];
  const float* eoutw = (const float*)d_in[14];
  const float* eoutb = (const float*)d_in[15];
  const float* elng  = (const float*)d_in[16];
  const float* elnb  = (const float*)d_in[17];
  const float* ed1w  = (const float*)d_in[18];
  const float* ed1b  = (const float*)d_in[19];
  const float* ed2w  = (const float*)d_in[20];
  const float* ed2b  = (const float*)d_in[21];
  const float* dq1w  = (const float*)d_in[22];
  const float* dq1b  = (const float*)d_in[23];
  const float* do1w  = (const float*)d_in[24];
  const float* do1b  = (const float*)d_in[25];
  const float* dq2w  = (const float*)d_in[26];
  const float* dq2b  = (const float*)d_in[27];
  const float* do2w  = (const float*)d_in[28];
  const float* do2b  = (const float*)d_in[29];
  const float* dlng  = (const float*)d_in[30];
  const float* dlnb  = (const float*)d_in[31];
  const float* dd1w  = (const float*)d_in[32];
  const float* dd1b  = (const float*)d_in[33];
  const float* dd2w  = (const float*)d_in[34];
  const float* dd2b  = (const float*)d_in[35];
  const float* finw  = (const float*)d_in[36];
  const float* finb  = (const float*)d_in[37];

  // ---- workspace carve-up (16B aligned sub-buffers) ----
  char* base = (char*)d_ws;
  size_t pos = 0;
  auto AF = [&](long n) { float* p = (float*)(base + pos);
                          pos += ((size_t)n * 4 + 15) & ~(size_t)15; return p; };
  auto AH = [&](long n) { _Float16* p = (_Float16*)(base + pos);
                          pos += ((size_t)n * 2 + 15) & ~(size_t)15; return p; };
  float* ENC = AF((long)ROWS * EMB);
  float* DEC = AF((long)ROWS * EMB);
  float* BIG = AF((long)ROWS * DFF);       // QKV / FFN-mid (f32 view)
  float* ATT = AF((long)ROWS * EMB);
  float* T1  = AF((long)ROWS * EMB);
  float* T2  = AF((long)ROWS * EMB);
  float* H1  = AF((long)ROWS * HID);
  float* H2  = AF((long)ROWS * FO);
  float* MS  = AF((long)ROWS * HID);
  float* CNT = AF(NN);
  // f16 activation mirrors
  _Float16* hXC  = AH((long)ROWS * FI);
  _Float16* hMS  = AH((long)ROWS * HID);
  _Float16* hH1  = AH((long)ROWS * HID);
  _Float16* hHS  = AH((long)ROWS * FO);
  _Float16* hENC = AH((long)ROWS * EMB);
  _Float16* hDEC = AH((long)ROWS * EMB);
  _Float16* hATT = AH((long)ROWS * EMB);
  _Float16* hT1  = AH((long)ROWS * EMB);
  _Float16* hT2  = AH((long)ROWS * EMB);
  _Float16* hBIG = AH((long)ROWS * DFF);
  // f16 weights, canonical [N][K] (k contiguous)
  _Float16* hWs1l = AH((long)HID * FI);
  _Float16* hWs1r = AH((long)HID * FI);
  _Float16* hWs2l = AH((long)FO * HID);
  _Float16* hWs2r = AH((long)FO * HID);
  _Float16* hWemb = AH((long)EMB * FO);
  _Float16* hWeqkv = AH(2L * 1536 * 512);
  _Float16* hWeout = AH(2L * 512 * 512);
  _Float16* hWed1  = AH(2L * DFF * EMB);
  _Float16* hWed2  = AH(2L * EMB * DFF);
  _Float16* hWdq1  = AH(2L * 1536 * 512);
  _Float16* hWdo1  = AH(2L * 512 * 512);
  _Float16* hWdq2  = AH(2L * 1536 * 512);
  _Float16* hWdo2  = AH(2L * 512 * 512);
  _Float16* hWdd1  = AH(2L * DFF * EMB);
  _Float16* hWdd2  = AH(2L * EMB * DFF);
  _Float16* hWfin  = AH(8L * FO * 8192);

  #define NB(n) ((unsigned)(((long)(n) + 255) / 256))
  auto GEMM = [&](const _Float16* Ah, long lda, const _Float16* Bh, long ldb,
                  float* C, long scm, long scn, _Float16* Ch, long ldch,
                  const float* bias, int M, int N, int K, int acc, int relu) {
    gemm_wmma<<<dim3((unsigned)(N / 64), (unsigned)(M / 128)), 256, 0, stream>>>(
        Ah, lda, Bh, ldb, C, scm, scn, Ch, ldch, bias, K, acc, relu);
  };

  // ---- one-time (per launch) f16 conversions ----
  cvt_kernel<<<NB((long)ROWS * FI), 256, 0, stream>>>(hXC, x, (long)ROWS * FI);
  cvtT_kernel<<<NB(FI * HID), 256, 0, stream>>>(hWs1l, s1wl, FI, HID);
  cvtT_kernel<<<NB(FI * HID), 256, 0, stream>>>(hWs1r, s1wr, FI, HID);
  cvtT_kernel<<<NB(HID * FO), 256, 0, stream>>>(hWs2l, s2wl, HID, FO);
  cvtT_kernel<<<NB(HID * FO), 256, 0, stream>>>(hWs2r, s2wr, HID, FO);
  cvtT_kernel<<<NB(FO * EMB), 256, 0, stream>>>(hWemb, embw, FO, EMB);
  cvt_kernel<<<NB(2L * 1536 * 512), 256, 0, stream>>>(hWeqkv, eqkvw, 2L * 1536 * 512);
  cvt_kernel<<<NB(2L * 512 * 512), 256, 0, stream>>>(hWeout, eoutw, 2L * 512 * 512);
  cvt_kernel<<<NB(2L * 1536 * 512), 256, 0, stream>>>(hWdq1, dq1w, 2L * 1536 * 512);
  cvt_kernel<<<NB(2L * 512 * 512), 256, 0, stream>>>(hWdo1, do1w, 2L * 512 * 512);
  cvt_kernel<<<NB(2L * 1536 * 512), 256, 0, stream>>>(hWdq2, dq2w, 2L * 1536 * 512);
  cvt_kernel<<<NB(2L * 512 * 512), 256, 0, stream>>>(hWdo2, do2w, 2L * 512 * 512);
  for (int i = 0; i < 2; ++i) {
    cvtT_kernel<<<NB((long)EMB * DFF), 256, 0, stream>>>(hWed1 + (long)i * DFF * EMB,
                                                         ed1w + (long)i * EMB * DFF, EMB, DFF);
    cvtT_kernel<<<NB((long)EMB * DFF), 256, 0, stream>>>(hWed2 + (long)i * EMB * DFF,
                                                         ed2w + (long)i * DFF * EMB, DFF, EMB);
    cvtT_kernel<<<NB((long)EMB * DFF), 256, 0, stream>>>(hWdd1 + (long)i * DFF * EMB,
                                                         dd1w + (long)i * EMB * DFF, EMB, DFF);
    cvtT_kernel<<<NB((long)EMB * DFF), 256, 0, stream>>>(hWdd2 + (long)i * EMB * DFF,
                                                         dd2w + (long)i * DFF * EMB, DFF, EMB);
  }
  for (int p = 0; p < 8; ++p)
    cvtT_kernel<<<NB(8192L * 64), 256, 0, stream>>>(hWfin + (long)p * FO * 8192,
                                                    finw + (long)p * 8192 * FO, 8192, FO);

  // ---- SAGE layer 1 ----
  zero_kernel<<<NB((long)ROWS * FI), 256, 0, stream>>>(MS, (long)ROWS * FI);
  zero_kernel<<<NB(NN), 256, 0, stream>>>(CNT, NN);
  count_kernel<<<NB(NE), 256, 0, stream>>>(edst, CNT, NE);
  scatter_kernel<<<NB((long)NE * SEQ * FI), 256, 0, stream>>>(x, esrc, edst, MS, NE, SEQ * FI);
  mean_kernel<<<NB((long)ROWS * FI), 256, 0, stream>>>(MS, hMS, CNT, (long)ROWS * FI, SEQ * FI);
  GEMM(hMS, FI, hWs1l, FI, H1, HID, 1, hH1, HID, s1b,    ROWS, HID, FI, 0, 0);
  GEMM(hXC, FI, hWs1r, FI, H1, HID, 1, hH1, HID, nullptr, ROWS, HID, FI, 1, 1);

  // ---- SAGE layer 2 ----
  zero_kernel<<<NB((long)ROWS * HID), 256, 0, stream>>>(MS, (long)ROWS * HID);
  scatter_kernel<<<NB((long)NE * SEQ * HID), 256, 0, stream>>>(H1, esrc, edst, MS, NE, SEQ * HID);
  mean_kernel<<<NB((long)ROWS * HID), 256, 0, stream>>>(MS, hMS, CNT, (long)ROWS * HID, SEQ * HID);
  GEMM(hMS, HID, hWs2l, HID, H2, FO, 1, nullptr, 0, s2b,    ROWS, FO, HID, 0, 0);
  GEMM(hH1, HID, hWs2r, HID, H2, FO, 1, nullptr, 0, nullptr, ROWS, FO, HID, 1, 1);

  // ---- sLSTM scan (node dim = time) ----
  slstm_kernel<<<1, 256, 0, stream>>>(H2, slW, slB, hHS);

  // ---- embedding, enc/dec init ----
  GEMM(hHS, FO, hWemb, FO, ENC, EMB, 1, hENC, EMB, embb, ROWS, EMB, FO, 0, 0);
  copy2_kernel<<<NB((long)ROWS * EMB), 256, 0, stream>>>(DEC, hDEC, ENC, hENC, (long)ROWS * EMB);

  // ---- encoder blocks ----
  for (int i = 0; i < 2; ++i) {
    GEMM(hENC, EMB, hWeqkv + (long)i * 1536 * 512, EMB, BIG, 1536, 1, hBIG, 1536,
         eqkvb + (long)i * 1536, ROWS, 1536, EMB, 0, 0);
    flash_attn<<<dim3(8, 128), 128, 0, stream>>>(hBIG, ATT, hATT, 0.125f);
    GEMM(hATT, EMB, hWeout + (long)i * 512 * 512, EMB, T1, 512, 1, nullptr, 0,
         eoutb + (long)i * 512, ROWS, EMB, EMB, 0, 0);
    add_ln<<<ROWS, 256, 0, stream>>>(ENC, T1, elng + (i * 2 + 0) * 512, elnb + (i * 2 + 0) * 512, T2, hT2);
    GEMM(hT2, EMB, hWed1 + (long)i * DFF * EMB, EMB, BIG, DFF, 1, hBIG, DFF,
         ed1b + (long)i * DFF, ROWS, DFF, EMB, 0, 0);
    GEMM(hBIG, DFF, hWed2 + (long)i * EMB * DFF, DFF, ATT, EMB, 1, nullptr, 0,
         ed2b + (long)i * EMB, ROWS, EMB, DFF, 0, 0);
    add_ln<<<ROWS, 256, 0, stream>>>(T2, ATT, elng + (i * 2 + 1) * 512, elnb + (i * 2 + 1) * 512, ENC, hENC);
  }

  // ---- decoder blocks ----
  for (int i = 0; i < 2; ++i) {
    // self attention
    GEMM(hDEC, EMB, hWdq1 + (long)i * 1536 * 512, EMB, BIG, 1536, 1, hBIG, 1536,
         dq1b + (long)i * 1536, ROWS, 1536, EMB, 0, 0);
    flash_attn<<<dim3(8, 128), 128, 0, stream>>>(hBIG, ATT, hATT, 0.125f);
    GEMM(hATT, EMB, hWdo1 + (long)i * 512 * 512, EMB, T1, 512, 1, nullptr, 0,
         do1b + (long)i * 512, ROWS, EMB, EMB, 0, 0);
    add_ln<<<ROWS, 256, 0, stream>>>(DEC, T1, dlng + (i * 4 + 0) * 512, dlnb + (i * 4 + 0) * 512, T2, hT2); // o1
    // cross attention: Q from o1, K/V from enc
    GEMM(hT2, EMB, hWdq2 + (long)i * 1536 * 512, EMB, BIG, 1536, 1, hBIG, 1536,
         dq2b + (long)i * 1536, ROWS, 512, EMB, 0, 0);
    GEMM(hENC, EMB, hWdq2 + (long)i * 1536 * 512 + (long)512 * 512, EMB, BIG + 512, 1536, 1,
         hBIG + 512, 1536, dq2b + (long)i * 1536 + 512, ROWS, 1024, EMB, 0, 0);
    flash_attn<<<dim3(8, 128), 128, 0, stream>>>(hBIG, ATT, hATT, 0.125f);
    GEMM(hATT, EMB, hWdo2 + (long)i * 512 * 512, EMB, T1, 512, 1, nullptr, 0,
         do2b + (long)i * 512, ROWS, EMB, EMB, 0, 0);
    add_ln<<<ROWS, 256, 0, stream>>>(T2, T1, dlng + (i * 4 + 1) * 512, dlnb + (i * 4 + 1) * 512, T1, hT1); // o2
    // FFN #1
    GEMM(hT1, EMB, hWdd1 + (long)i * DFF * EMB, EMB, BIG, DFF, 1, hBIG, DFF,
         dd1b + (long)i * DFF, ROWS, DFF, EMB, 0, 0);
    GEMM(hBIG, DFF, hWdd2 + (long)i * EMB * DFF, DFF, T2, EMB, 1, nullptr, 0,
         dd2b + (long)i * EMB, ROWS, EMB, DFF, 0, 0);
    add_ln<<<ROWS, 256, 0, stream>>>(T1, T2, dlng + (i * 4 + 2) * 512, dlnb + (i * 4 + 2) * 512, T2, hT2); // o3
    // FFN #2 (same weights, as in source)
    GEMM(hT2, EMB, hWdd1 + (long)i * DFF * EMB, EMB, BIG, DFF, 1, hBIG, DFF,
         dd1b + (long)i * DFF, ROWS, DFF, EMB, 0, 0);
    GEMM(hBIG, DFF, hWdd2 + (long)i * EMB * DFF, DFF, T1, EMB, 1, nullptr, 0,
         dd2b + (long)i * EMB, ROWS, EMB, DFF, 0, 0);
    add_ln<<<ROWS, 256, 0, stream>>>(T2, T1, dlng + (i * 4 + 3) * 512, dlnb + (i * 4 + 3) * 512, DEC, hDEC);
  }

  // ---- final projection: out[o][n][p] = flat[n,:] . fin_w[p][:,o] + fin_b[p][o]
  // flat[n][d] = DEC[n*8192 + d]  ->  hDEC viewed as [512][8192] f16
  float* out = (float*)d_out;
  for (int p = 0; p < 8; ++p) {
    GEMM(hDEC, 8192, hWfin + (long)p * FO * 8192, 8192,
         out + p, 8, (long)NN * 8, nullptr, 0, finb + (long)p * FO, NN, FO, 8192, 0, 0);
  }
}
